// MultiHeadAttention_57569741635718
// MI455X (gfx1250) — matrix operand order repaired and checked by reading, not compiled, not run on hardware
//
#include <hip/hip_runtime.h>
#include <stdint.h>

typedef unsigned short u16;
typedef __attribute__((ext_vector_type(16))) __bf16 bf16x16;
typedef __attribute__((ext_vector_type(8)))  __bf16 bf16x8;
typedef __attribute__((ext_vector_type(8)))  float  floatx8;
typedef __attribute__((ext_vector_type(4)))  float  floatx4;
typedef __attribute__((ext_vector_type(8)))  u16    ushort8;

#define DEV_INLINE __device__ __forceinline__

constexpr int kB  = 4;
constexpr int kS  = 2048;
constexpr int kD  = 1024;
constexpr int kH  = 16;
constexpr int kHD = 64;
constexpr int kM  = kB * kS;   // 8192
constexpr int kBH = kB * kH;   // 64

// ---------------- fp32 -> bf16 (RNE) ----------------
DEV_INLINE u16 f32_bf16(float f) {
  union { float f; unsigned u; } c{f};
  unsigned u = c.u;
  u += 0x7FFFu + ((u >> 16) & 1u);
  return (u16)(u >> 16);
}
DEV_INLINE u16 to_bf16(float f) { return f32_bf16(f); }
DEV_INLINE u16 to_bf16(u16 x)   { return x; }

// 8 contiguous source elems -> 8 bf16 bits, as one 16B LDS store
DEV_INLINE ushort8 cvt8(const float* __restrict__ p) {
  floatx4 a = *(const floatx4*)p;
  floatx4 b = *(const floatx4*)(p + 4);
  ushort8 o;
#pragma unroll
  for (int i = 0; i < 4; ++i) { o[i] = f32_bf16(a[i]); o[4 + i] = f32_bf16(b[i]); }
  return o;
}

// ---------------- CDNA5 async global -> LDS (ASYNCcnt path) ----------------
// 16B per lane, LDS byte address in a VGPR, 64-bit global address in a VGPR pair.
DEV_INLINE void async_g2l_b128(unsigned lds_addr, const u16* gptr) {
  asm volatile("global_load_async_to_lds_b128 %0, %1, off"
               :: "v"(lds_addr), "v"(gptr)
               : "memory");
}
DEV_INLINE void wait_async0() {
  asm volatile("s_wait_asynccnt 0" ::: "memory");
}

// ---------------- LDS staging ----------------
// Copy ROWS x BK fp32 tile (K contiguous in memory) into LDS [row][BKP] as bf16.
template<int ROWS, int BK, int BKP>
DEV_INLINE void stage_k_major_f32(u16* dst, const float* __restrict__ src, int ld, int tid) {
  constexpr int KV   = BK / 8;
  constexpr int VECS = ROWS * KV;
#pragma unroll
  for (int v = tid; v < VECS; v += 256) {
    int row = v / KV, kc = (v % KV) * 8;
    *(ushort8*)(dst + row * BKP + kc) = cvt8(src + (size_t)row * ld + kc);
  }
}
// bf16 source: pure byte copy -> async DMA into LDS, no VGPR round-trip.
// Caller must wait_async0() before the publishing __syncthreads().
template<int ROWS, int BK, int BKP>
DEV_INLINE void stage_k_major_async(u16* dst, const u16* __restrict__ src, int ld, int tid) {
  constexpr int KV   = BK / 8;
  constexpr int VECS = ROWS * KV;   // always a multiple of 256 here
#pragma unroll
  for (int v = tid; v < VECS; v += 256) {
    int row = v / KV, kc = (v % KV) * 8;
    async_g2l_b128((unsigned)(uintptr_t)(dst + row * BKP + kc),
                   src + (size_t)row * ld + kc);
  }
}
// Stage B (row-major K x N) transposed into LDS as [n][BKP].
template<int BN, int BK, int BKP, typename T>
DEV_INLINE void stage_transpose(u16* dst, const T* __restrict__ src, int ld, int tid) {
  constexpr int TOT = BN * BK;
  for (int e = tid; e < TOT; e += 256) {
    int k = e / BN, n = e % BN;
    dst[n * BKP + k] = to_bf16(src[(size_t)k * ld + n]);
  }
}

// ---------------- WMMA fragment loads (ISA §7.12.2 layouts) ----------------
// A 16x32 bf16: lane<16 holds K {half*8..+7, 16+half*8..+7}
DEV_INLINE bf16x16 load_frag_a(const u16* p, int half) {
  bf16x8 a = *(const bf16x8*)(p + half * 8);
  bf16x8 b = *(const bf16x8*)(p + 16 + half * 8);
  bf16x16 r;
#pragma unroll
  for (int i = 0; i < 8; ++i) { r[i] = a[i]; r[i + 8] = b[i]; }
  return r;
}
// B 32x16 bf16: lane<16 holds K {half*8..+7, 16+half*8..+7} (same as A)
DEV_INLINE bf16x16 load_frag_b(const u16* p, int half) {
  bf16x8 a = *(const bf16x8*)(p + half * 8);
  bf16x8 b = *(const bf16x8*)(p + 16 + half * 8);
  bf16x16 r;
#pragma unroll
  for (int i = 0; i < 8; ++i) { r[i] = a[i]; r[i + 8] = b[i]; }
  return r;
}

// ---------------- wave-level MMA over one LDS stage ----------------
template<int BM, int BN, int BK, int BKP, int WR, int WC>
DEV_INLINE void wave_mma(const u16* At, const u16* Bt,
                         floatx8 (*acc)[(BN / WC) / 16], int wid, int lane) {
  constexpr int MT = (BM / WR) / 16, NT = (BN / WC) / 16;
  const int wrow = wid / WC, wcol = wid % WC;
  const int m0 = wrow * (BM / WR), n0 = wcol * (BN / WC);
  const int r16 = lane & 15, half = (lane >> 4) & 1;
#pragma unroll
  for (int kk = 0; kk < BK; kk += 32) {
    bf16x16 af[MT], bfv[NT];
#pragma unroll
    for (int i = 0; i < MT; ++i)
      af[i] = load_frag_a(At + (m0 + i * 16 + r16) * BKP + kk, half);
#pragma unroll
    for (int j = 0; j < NT; ++j)
      bfv[j] = load_frag_b(Bt + (n0 + j * 16 + r16) * BKP + kk, half);
#pragma unroll
    for (int i = 0; i < MT; ++i)
#pragma unroll
      for (int j = 0; j < NT; ++j)
        acc[i][j] = __builtin_amdgcn_wmma_f32_16x16x32_bf16(
            false, af[i], false, bfv[j], (short)0, acc[i][j], false, false);
  }
}

// ---------------- projection GEMMs ----------------
// QKV=true : A = fp32 [8192,1024], B = W fp32 [1024,1024]; +bias; bf16 out
//            scattered to [B,H,S,64].
// QKV=false: A = ctx bf16 [8192,1024] (async-staged), B = Wo fp32; +bias;
//            fp32 out [8192,1024].
template<bool QKV>
__global__ __launch_bounds__(256) void k_proj(const void* __restrict__ Ap,
                                              const float* __restrict__ W,
                                              const float* __restrict__ bias,
                                              void* __restrict__ Cp) {
  constexpr int BM = 128, BN = 128, BK = 64, BKP = 72, WR = 2, WC = 4;
  constexpr int MT = (BM / WR) / 16, NT = (BN / WC) / 16;
  __shared__ u16 At[BM * BKP];
  __shared__ u16 Bt[BN * BKP];
  const int tid = threadIdx.x, wid = tid >> 5, lane = tid & 31;
  const int bm = blockIdx.y * BM, bn = blockIdx.x * BN;

  floatx8 acc[MT][NT];
  const floatx8 z = {0.f, 0.f, 0.f, 0.f, 0.f, 0.f, 0.f, 0.f};
#pragma unroll
  for (int i = 0; i < MT; ++i)
#pragma unroll
    for (int j = 0; j < NT; ++j) acc[i][j] = z;

  for (int k0 = 0; k0 < kD; k0 += BK) {
    if constexpr (QKV) {
      stage_k_major_f32<BM, BK, BKP>(At, (const float*)Ap + (size_t)bm * kD + k0, kD, tid);
    } else {
      stage_k_major_async<BM, BK, BKP>(At, (const u16*)Ap + (size_t)bm * kD + k0, kD, tid);
    }
    stage_transpose<BN, BK, BKP>(Bt, W + (size_t)k0 * kD + bn, kD, tid);
    if constexpr (!QKV) wait_async0();
    __syncthreads();
    wave_mma<BM, BN, BK, BKP, WR, WC>(At, Bt, acc, wid, lane);
    __syncthreads();
  }

  const int wrow = wid / WC, wcol = wid % WC, half = lane >> 4, r16 = lane & 15;
#pragma unroll
  for (int i = 0; i < MT; ++i)
#pragma unroll
    for (int j = 0; j < NT; ++j)
#pragma unroll
      for (int r = 0; r < 8; ++r) {
        int gm = bm + wrow * (BM / WR) + i * 16 + half * 8 + r;
        int gn = bn + wcol * (BN / WC) + j * 16 + r16;
        float v = acc[i][j][r] + bias[gn];
        if constexpr (QKV) {
          int b = gm >> 11, s = gm & (kS - 1);
          int h = gn >> 6, dd = gn & (kHD - 1);
          ((u16*)Cp)[(((size_t)b * kH + h) * kS + s) * kHD + dd] = f32_bf16(v);
        } else {
          ((float*)Cp)[(size_t)gm * kD + gn] = v;
        }
      }
}

// ---------------- scores: per head, S x S = Q (S x 64) * K^T ----------------
__global__ __launch_bounds__(256) void k_scores(const u16* __restrict__ qb,
                                                const u16* __restrict__ kb,
                                                float* __restrict__ attn) {
  constexpr int BM = 128, BN = 128, BK = 64, BKP = 72, WR = 2, WC = 4;
  constexpr int MT = (BM / WR) / 16, NT = (BN / WC) / 16;
  __shared__ u16 At[BM * BKP];
  __shared__ u16 Bt[BN * BKP];
  const int tid = threadIdx.x, wid = tid >> 5, lane = tid & 31;
  const int bh = blockIdx.z;
  const int bm = blockIdx.y * BM, bn = blockIdx.x * BN;
  const u16* A  = qb + (size_t)bh * kS * kHD;
  const u16* Bm = kb + (size_t)bh * kS * kHD;

  floatx8 acc[MT][NT];
  const floatx8 z = {0.f, 0.f, 0.f, 0.f, 0.f, 0.f, 0.f, 0.f};
#pragma unroll
  for (int i = 0; i < MT; ++i)
#pragma unroll
    for (int j = 0; j < NT; ++j) acc[i][j] = z;

  // K-dim = head dim = 64, contiguous in both operands (NT gemm): one stage,
  // both tiles are already bf16 -> async DMA into LDS.
  stage_k_major_async<BM, BK, BKP>(At, A + (size_t)bm * kHD, kHD, tid);
  stage_k_major_async<BN, BK, BKP>(Bt, Bm + (size_t)bn * kHD, kHD, tid);
  wait_async0();
  __syncthreads();
  wave_mma<BM, BN, BK, BKP, WR, WC>(At, Bt, acc, wid, lane);

  float* C = attn + (size_t)bh * kS * kS;
  const int wrow = wid / WC, wcol = wid % WC, half = lane >> 4, r16 = lane & 15;
#pragma unroll
  for (int i = 0; i < MT; ++i)
#pragma unroll
    for (int j = 0; j < NT; ++j)
#pragma unroll
      for (int r = 0; r < 8; ++r) {
        int gm = bm + wrow * (BM / WR) + i * 16 + half * 8 + r;
        int gn = bn + wcol * (BN / WC) + j * 16 + r16;
        C[(size_t)gm * kS + gn] = acc[i][j][r] * 0.125f;  // 1/sqrt(64)
      }
}

// ---------------- softmax: one 256-thread block per row of 2048 ----------------
__global__ __launch_bounds__(256) void k_softmax(float* __restrict__ attn) {
  const size_t row = blockIdx.x;
  float* p = attn + row * (size_t)kS;
  const int tid = threadIdx.x, wid = tid >> 5, lane = tid & 31;
  __shared__ float red[8];

  float v[8];
  float mx = -3.402823466e38f;
#pragma unroll
  for (int i = 0; i < 8; ++i) { v[i] = p[tid + i * 256]; mx = fmaxf(mx, v[i]); }
#pragma unroll
  for (int off = 16; off > 0; off >>= 1) mx = fmaxf(mx, __shfl_xor(mx, off, 32));
  if (lane == 0) red[wid] = mx;
  __syncthreads();
  mx = red[0];
#pragma unroll
  for (int i = 1; i < 8; ++i) mx = fmaxf(mx, red[i]);

  float s = 0.f;
#pragma unroll
  for (int i = 0; i < 8; ++i) { v[i] = __expf(v[i] - mx); s += v[i]; }
#pragma unroll
  for (int off = 16; off > 0; off >>= 1) s += __shfl_xor(s, off, 32);
  __syncthreads();
  if (lane == 0) red[wid] = s;
  __syncthreads();
  s = red[0];
#pragma unroll
  for (int i = 1; i < 8; ++i) s += red[i];

  const float inv = 1.0f / s;
#pragma unroll
  for (int i = 0; i < 8; ++i) p[tid + i * 256] = v[i] * inv;
}

// ---------------- ctx: per head, (S x S fp32 probs) * V (S x 64 bf16) ----------------
__global__ __launch_bounds__(256) void k_ctx(const float* __restrict__ attn,
                                             const u16* __restrict__ vb,
                                             u16* __restrict__ ctxb) {
  constexpr int BM = 128, BN = 64, BK = 64, BKP = 72, WR = 4, WC = 2;
  constexpr int MT = (BM / WR) / 16, NT = (BN / WC) / 16;
  __shared__ u16 At[BM * BKP];
  __shared__ u16 Bt[BN * BKP];
  const int tid = threadIdx.x, wid = tid >> 5, lane = tid & 31;
  const int bh = blockIdx.z;
  const int bm = blockIdx.y * BM, bn = blockIdx.x * BN;  // bn == 0
  const float* A  = attn + (size_t)bh * kS * kS;
  const u16*   Bv = vb + (size_t)bh * kS * kHD;

  floatx8 acc[MT][NT];
  const floatx8 z = {0.f, 0.f, 0.f, 0.f, 0.f, 0.f, 0.f, 0.f};
#pragma unroll
  for (int i = 0; i < MT; ++i)
#pragma unroll
    for (int j = 0; j < NT; ++j) acc[i][j] = z;

  for (int k0 = 0; k0 < kS; k0 += BK) {
    stage_k_major_f32<BM, BK, BKP>(At, A + (size_t)bm * kS + k0, kS, tid);
    stage_transpose<BN, BK, BKP>(Bt, Bv + (size_t)k0 * kHD + bn, kHD, tid);
    __syncthreads();
    wave_mma<BM, BN, BK, BKP, WR, WC>(At, Bt, acc, wid, lane);
    __syncthreads();
  }

  const int b = bh / kH, h = bh % kH;
  const int wrow = wid / WC, wcol = wid % WC, half = lane >> 4, r16 = lane & 15;
#pragma unroll
  for (int i = 0; i < MT; ++i)
#pragma unroll
    for (int j = 0; j < NT; ++j)
#pragma unroll
      for (int r = 0; r < 8; ++r) {
        int gm = bm + wrow * (BM / WR) + i * 16 + half * 8 + r;
        int gn = bn + wcol * (BN / WC) + j * 16 + r16;
        ctxb[((size_t)b * kS + gm) * kD + h * kHD + gn] = f32_bf16(acc[i][j][r]);
      }
}

// ---------------- launch ----------------
extern "C" void kernel_launch(void* const* d_in, const int* in_sizes, int n_in,
                              void* d_out, int out_size, void* d_ws, size_t ws_size,
                              hipStream_t stream) {
  (void)in_sizes; (void)n_in; (void)out_size; (void)ws_size;
  const float* q  = (const float*)d_in[0];
  const float* k  = (const float*)d_in[1];
  const float* v  = (const float*)d_in[2];
  const float* Wq = (const float*)d_in[3];
  const float* bq = (const float*)d_in[4];
  const float* Wk = (const float*)d_in[5];
  const float* bk = (const float*)d_in[6];
  const float* Wv = (const float*)d_in[7];
  const float* bv = (const float*)d_in[8];
  const float* Wo = (const float*)d_in[9];
  const float* bo = (const float*)d_in[10];

  float* out  = (float*)d_out;
  float* attn = out + (size_t)kM * kD;  // out [B,S,D] then attn [B,H,S,S]

  const size_t elems = (size_t)kM * kD;  // == B*H*S*64
  u16* qb   = (u16*)d_ws;
  u16* kbuf = qb + elems;
  u16* vbuf = kbuf + elems;
  u16* ctxb = vbuf + elems;

  dim3 blk(256);
  dim3 gproj(kD / 128, kM / 128, 1);     // 8 x 64
  k_proj<true><<<gproj, blk, 0, stream>>>(q, Wq, bq, qb);
  k_proj<true><<<gproj, blk, 0, stream>>>(k, Wk, bk, kbuf);
  k_proj<true><<<gproj, blk, 0, stream>>>(v, Wv, bv, vbuf);

  dim3 gsc(kS / 128, kS / 128, kBH);     // 16 x 16 x 64
  k_scores<<<gsc, blk, 0, stream>>>(qb, kbuf, attn);

  k_softmax<<<dim3(kBH * kS), blk, 0, stream>>>(attn);  // 131072 rows

  dim3 gctx(1, kS / 128, kBH);           // 1 x 16 x 64
  k_ctx<<<gctx, blk, 0, stream>>>(attn, vbuf, ctxb);

  k_proj<false><<<gproj, blk, 0, stream>>>(ctxb, Wo, bo, out);
}
